// PINN_10780367913688
// MI455X (gfx1250) — compile-verified, matrix operand order checked by reading, not run on hardware
//
#include <hip/hip_runtime.h>
#include <hip/hip_bf16.h>

// ---- WMMA vector types (gfx1250, wave32) ----
typedef __attribute__((ext_vector_type(16))) __bf16 v16bf;
typedef __attribute__((ext_vector_type(8)))  float  v8f;
typedef __attribute__((ext_vector_type(4)))  unsigned int u32x4;

#define WAVES   8
#define BLOCK   (WAVES * 32)
#define GRID    512
#define PPW     2            // points per wave per iteration (share B fragments)
#define HID     128
#define EMB     64
#define NLAYERS 4

// cfg constants
#define CSE  0.5f
#define CLE  0.00574f
#define AC1  2.0f
#define AC2  1.0f
#define AC3  0.01f
#define TWO_PI 6.283185307179586f

__device__ __forceinline__ unsigned short f2bf(float f) {
    unsigned int u = __float_as_uint(f);
    unsigned int r = u + 0x7FFFu + ((u >> 16) & 1u);   // RNE
    return (unsigned short)(r >> 16);
}
__device__ __forceinline__ float bf2f(unsigned short h) {
    return __uint_as_float(((unsigned int)h) << 16);
}

// Branch-free tanh: 1 - 2/(exp(2x)+1).  v_exp_f32 + v_rcp_f32, no EXEC divergence.
__device__ __forceinline__ float fast_tanh(float x) {
    float e = __expf(2.0f * x);                 // inf for large x -> rcp -> 0 -> tanh -> 1
    return 1.0f - 2.0f * __builtin_amdgcn_rcpf(e + 1.0f);
}

// Load one 16x32 bf16 WMMA fragment (A or B) from an LDS row of 128 bf16.
// Lane layout (ISA 7.12.2): element j -> K = 16*(j/8) + 8*half + (j%8)
// => two contiguous 8x bf16 (16B) chunks per lane.
__device__ __forceinline__ v16bf load_frag(const unsigned short* row, int half, int k0) {
    union { u32x4 q[2]; v16bf v; } u;
    u.q[0] = *(const u32x4*)(row + k0 + 8 * half);
    u.q[1] = *(const u32x4*)(row + k0 + 16 + 8 * half);
    return u.v;
}

__global__ __launch_bounds__(BLOCK)
void PINN_ac_residual_kernel(const float* __restrict__ xin,   // (N,2)
                             const float* __restrict__ tin,   // (N,1)
                             const float* __restrict__ Bff,   // (3,EMB)
                             const float* __restrict__ Wh,    // (L,HID,HID)
                             const float* __restrict__ bh,    // (L,HID)
                             const float* __restrict__ Wout,  // (HID,2)
                             const float* __restrict__ bout,  // (2,)
                             float* __restrict__ out, int N)
{
    // ---- LDS staging ----
    __shared__ __align__(16) unsigned short sWt[NLAYERS][HID][HID];   // W^T bf16 (128 KB)
    __shared__ __align__(16) unsigned short sH[WAVES][PPW][16][HID];  // activation tiles (64 KB)
    __shared__ float sB[3][EMB];
    __shared__ float sBh[NLAYERS][HID];
    __shared__ float sWo[HID][2];
    __shared__ float sBo[2];
    __shared__ float sRed[WAVES][32];

    const int tid  = threadIdx.x;
    const int lane = tid & 31;
    const int w    = tid >> 5;

    // Stage W^T as bf16: sWt[l][n][k] = W[l][k][n]
    for (int idx = tid; idx < NLAYERS * HID * HID; idx += BLOCK) {
        int l = idx >> 14, rem = idx & 16383;
        int n = rem >> 7, k = rem & 127;
        sWt[l][n][k] = f2bf(Wh[((l * HID) + k) * HID + n]);
    }
    for (int idx = tid; idx < 3 * EMB; idx += BLOCK)       sB[idx / EMB][idx % EMB] = Bff[idx];
    for (int idx = tid; idx < NLAYERS * HID; idx += BLOCK) sBh[idx / HID][idx % HID] = bh[idx];
    for (int idx = tid; idx < HID * 2; idx += BLOCK)       sWo[idx >> 1][idx & 1] = Wout[idx];
    if (tid < 2) sBo[tid] = bout[tid];
    // Zero activation tiles once (pad rows 6,7,14,15 stay zero forever)
    {
        u32x4 z = (u32x4){0u, 0u, 0u, 0u};
        u32x4* ptr = (u32x4*)&sH[0][0][0][0];
        for (int idx = tid; idx < WAVES * PPW * 16 * HID / 8; idx += BLOCK) ptr[idx] = z;
    }
    __syncthreads();

    const int m    = lane & 15;   // WMMA row (A) / column (B,C,D) index
    const int half = lane >> 4;   // lane half: K sub-range (A/B) / row group (C/D)

    const int gw     = blockIdx.x * WAVES + w;
    const int stride = gridDim.x * WAVES * PPW;

    for (int p0 = gw * PPW; p0 < N; p0 += stride) {
        // ---- Fourier jet for both points -> sH rows 0..5 (pass0) / 8..13 (pass1) ----
#pragma unroll
        for (int pt = 0; pt < PPW; ++pt) {
            const int p  = p0 + pt;
            const float x0 = xin[2 * p + 0];
            const float x1 = xin[2 * p + 1];
            const float tt = tin[p];
            const int pass = half;               // lanes 0-15: pass 0; 16-31: mirrored pass
            const int cb   = m * 8;
            const float z0 = pass ? -x0 : x0;
            float ch[6][8];
#pragma unroll
            for (int q = 0; q < 8; ++q) {
                int col = cb + q;
                int j = col & 63;
                bool isSin = col < EMB;
                float b0 = sB[0][j], b1 = sB[1][j], b2 = sB[2][j];
                float f  = TWO_PI * (z0 * b0 + x1 * b1 + tt * b2);
                float g0 = TWO_PI * b0, g1 = TWO_PI * b1, g2 = TWO_PI * b2;
                float sn = __sinf(f), cs = __cosf(f);
                float v  = isSin ? sn : cs;      // value
                float dv = isSin ? cs : -sn;     // d/df
                float d2 = -v;                   // d2/df2
                ch[0][q] = v;
                ch[1][q] = dv * g2;              // d/dt
                ch[2][q] = dv * g0;              // d/dz0
                ch[3][q] = dv * g1;              // d/dz1
                ch[4][q] = d2 * g0 * g0;         // d2/dz0^2
                ch[5][q] = d2 * g1 * g1;         // d2/dz1^2
            }
#pragma unroll
            for (int c = 0; c < 6; ++c) {
                union { unsigned short s[8]; u32x4 q; } pk;
#pragma unroll
                for (int q = 0; q < 8; ++q) pk.s[q] = f2bf(ch[c][q]);
                *(u32x4*)&sH[w][pt][pass * 8 + c][cb] = pk.q;
            }
        }

        // ---- 4 hidden layers ----
#pragma unroll 1
        for (int l = 0; l < NLAYERS; ++l) {
            // A fragments for both points, all K steps, held in VGPRs.
            v16bf A[PPW][4];
#pragma unroll
            for (int pt = 0; pt < PPW; ++pt) {
                const unsigned short* arow = &sH[w][pt][m][0];
#pragma unroll
                for (int kt = 0; kt < 4; ++kt) A[pt][kt] = load_frag(arow, half, 32 * kt);
            }

#pragma unroll 2
            for (int nt = 0; nt < 8; ++nt) {
                const int n = nt * 16 + m;
                const unsigned short* wrow = &sWt[l][n][0];
                // B fragments loaded once, shared by both points.
                v16bf B0 = load_frag(wrow, half, 0);
                v16bf B1 = load_frag(wrow, half, 32);
                v16bf B2 = load_frag(wrow, half, 64);
                v16bf B3 = load_frag(wrow, half, 96);
                const float bias = sBh[l][n];
#pragma unroll
                for (int pt = 0; pt < PPW; ++pt) {
                    v8f acc = (v8f){0.f, 0.f, 0.f, 0.f, 0.f, 0.f, 0.f, 0.f};
                    acc = __builtin_amdgcn_wmma_f32_16x16x32_bf16(false, A[pt][0], false, B0, (short)0, acc, false, false);
                    acc = __builtin_amdgcn_wmma_f32_16x16x32_bf16(false, A[pt][1], false, B1, (short)0, acc, false, false);
                    acc = __builtin_amdgcn_wmma_f32_16x16x32_bf16(false, A[pt][2], false, B2, (short)0, acc, false, false);
                    acc = __builtin_amdgcn_wmma_f32_16x16x32_bf16(false, A[pt][3], false, B3, (short)0, acc, false, false);

                    // Per-lane tanh jet epilogue (acc[c] = channel c of this lane's unit)
                    float T  = fast_tanh(acc[0] + bias);
                    float s  = 1.f - T * T;        // tanh'
                    float sp = -2.f * T * s;       // tanh''
                    const int rb = half * 8;
                    sH[w][pt][rb + 0][n] = f2bf(T);
                    sH[w][pt][rb + 1][n] = f2bf(s * acc[1]);
                    sH[w][pt][rb + 2][n] = f2bf(s * acc[2]);
                    sH[w][pt][rb + 3][n] = f2bf(s * acc[3]);
                    sH[w][pt][rb + 4][n] = f2bf(sp * acc[2] * acc[2] + s * acc[4]);
                    sH[w][pt][rb + 5][n] = f2bf(sp * acc[3] * acc[3] + s * acc[5]);
                }
            }
        }

        // ---- output layer: 2 points x 2 passes x 7 dots = 28 lanes ----
        if (lane < PPW * 14) {
            int pt = lane / 14, r = lane % 14;
            int ps = r / 7, k = r % 7;
            int row = ps * 8 + (k < 6 ? k : 0);
            int oc  = (k < 6) ? 0 : 1;
            float acc = 0.f;
            const unsigned short* hrow = &sH[w][pt][row][0];
#pragma unroll
            for (int c8 = 0; c8 < HID / 8; ++c8) {
                union { u32x4 q; unsigned short s[8]; } blk;
                blk.q = *(const u32x4*)(hrow + c8 * 8);
#pragma unroll
                for (int q = 0; q < 8; ++q)
                    acc += bf2f(blk.s[q]) * sWo[c8 * 8 + q][oc];
            }
            if (k == 0) acc += sBo[0];
            if (k == 6) acc += sBo[1];
            sRed[w][lane] = acc;
        }

        // ---- lanes 0..PPW-1: combine passes into Allen-Cahn residual ----
        if (lane < PPW) {
            const float* rd = &sRed[w][lane * 14];
            float phi = 0.f, cc = 0.f, dphidt = 0.f, lap = 0.f;
#pragma unroll
            for (int ps = 0; ps < 2; ++ps) {
                float o0  = rd[ps * 7 + 0];
                float ot_ = rd[ps * 7 + 1];
                float ox0 = rd[ps * 7 + 2];
                float ox1 = rd[ps * 7 + 3];
                float oxx = rd[ps * 7 + 4];
                float oyy = rd[ps * 7 + 5];
                float o1  = rd[ps * 7 + 6];
                float T = fast_tanh(o0);
                float s = 1.f - T * T;
                float d2 = -2.f * T * s;
                float ph = 0.5f * T + 0.5f;
                phi    += 0.5f * ph;
                dphidt += 0.5f * (0.5f * s * ot_);
                lap    += 0.5f * 0.5f * (d2 * (ox0 * ox0 + ox1 * ox1) + s * (oxx + oyy));
                float cl  = (0.5f * fast_tanh(o1) + 0.5f) * (1.f - CSE + CLE);
                float hph = -2.f * ph * ph * ph + 3.f * ph * ph;
                cc += 0.5f * ((CSE - CLE) * hph + cl);
            }
            float h_phi = -2.f * phi * phi * phi + 3.f * phi * phi;
            float dh    = -6.f * phi * phi + 6.f * phi;
            float dg    =  4.f * phi * phi * phi - 6.f * phi * phi + 2.f * phi;
            float ac = dphidt
                     - AC1 * (cc - h_phi * (CSE - CLE) - CLE) * (CSE - CLE) * dh
                     + AC2 * dg
                     - AC3 * lap;
            int pp = p0 + lane;
            if (pp < N) out[pp] = ac;
        }
    }
}

extern "C" void kernel_launch(void* const* d_in, const int* in_sizes, int n_in,
                              void* d_out, int out_size, void* d_ws, size_t ws_size,
                              hipStream_t stream) {
    const float* x    = (const float*)d_in[0];
    const float* t    = (const float*)d_in[1];
    const float* Bff  = (const float*)d_in[2];
    const float* Wh   = (const float*)d_in[3];
    const float* bh   = (const float*)d_in[4];
    const float* Wout = (const float*)d_in[5];
    const float* bout = (const float*)d_in[6];
    float* out = (float*)d_out;
    const int N = in_sizes[1]; // t has N elements

    PINN_ac_residual_kernel<<<GRID, BLOCK, 0, stream>>>(x, t, Bff, Wh, bh, Wout, bout, out, N);
}
